// CustomLSTM_37443524887282
// MI455X (gfx1250) — matrix operand order repaired and checked by reading, not compiled, not run on hardware
//
#include <hip/hip_runtime.h>

// CustomLSTM fused kernel for gfx1250 (MI455X), wave32 + WMMA bf16.
//
// Layout convention (per CDNA5 ISA 7.12.2, wave32):
//  A 16x32 bf16 : lane L<16 holds row M=L, K = 0..7,16..23 ; lane L+16 holds
//                 row M=L, K = 8..15,24..31 (elem j of v16bf = j-th of that K set)
//  B 32x16 bf16 : lane L<16 holds col N=L, K = 0..7,16..23 ; lane L+16 col N=L, K=8..15,24..31
//  C/D 16x16 f32: lane L holds col N=L%16, VGPR v holds row M = v + 8*(L>=16)

#define T_LEN 5
#define F_IN  30
#define H_HID 20
#define GPAD  32                    // per-gate padded width
#define NPAD  128                   // 4 gates * GPAD
#define KPAD  32                    // padded K (>= F_IN, >= H_HID)
#define WAVES 8                     // waves per block
#define ROWS_PER_WAVE 16
#define ROWS_PER_BLOCK (WAVES * ROWS_PER_WAVE)

typedef __bf16 bf16_t;
typedef bf16_t v16bf __attribute__((ext_vector_type(16)));
typedef bf16_t v8bf  __attribute__((ext_vector_type(8)));
typedef float  v8f   __attribute__((ext_vector_type(8)));

static __device__ __forceinline__ float fast_rcp(float x) {
#if __has_builtin(__builtin_amdgcn_rcpf)
  return __builtin_amdgcn_rcpf(x);
#else
  return 1.0f / x;
#endif
}

static __device__ __forceinline__ float fast_tanh1(float x) {
#if __has_builtin(__builtin_amdgcn_tanhf)
  return __builtin_amdgcn_tanhf(x);
#elif __has_builtin(__builtin_amdgcn_tanh_f32)
  return __builtin_amdgcn_tanh_f32(x);
#else
  float e = __expf(-2.0f * x);
  return (1.0f - e) * fast_rcp(1.0f + e);
#endif
}

static __device__ __forceinline__ float fast_sigmoid(float x) {
  return fast_rcp(1.0f + __expf(-x));
}

static __device__ __forceinline__ v8f sig8(v8f x) {
  v8f r;
#pragma unroll
  for (int i = 0; i < 8; ++i) r[i] = fast_sigmoid(x[i]);
  return r;
}

static __device__ __forceinline__ v8f tanh8(v8f x) {
  v8f r;
#pragma unroll
  for (int i = 0; i < 8; ++i) r[i] = fast_tanh1(x[i]);
  return r;
}

// Load one 32x16 bf16 B-tile from transposed LDS weights ([n][k] rows, 64B).
static __device__ __forceinline__ v16bf load_btile(const bf16_t* __restrict__ base,
                                                   int n, int khalf) {
  const bf16_t* src = base + n * KPAD + khalf;      // 16B-aligned
  v8bf lo = *(const v8bf*)(src);
  v8bf hi = *(const v8bf*)(src + 16);
  v16bf r;
#pragma unroll
  for (int j = 0; j < 8; ++j) { r[j] = lo[j]; r[8 + j] = hi[j]; }
  return r;
}

__global__ __launch_bounds__(WAVES * 32, 1)
void lstm_fused_wmma(const float* __restrict__ x,
                     const float* __restrict__ wf, const float* __restrict__ wi,
                     const float* __restrict__ wo, const float* __restrict__ wc,
                     const float* __restrict__ uf, const float* __restrict__ ui,
                     const float* __restrict__ uo, const float* __restrict__ uc,
                     const float* __restrict__ bfp, const float* __restrict__ bip,
                     const float* __restrict__ bop, const float* __restrict__ bcp,
                     float* __restrict__ out, int Btot)
{
  // Transposed, zero-padded bf16 weights in LDS: [n][k], n = gate*32 + h.
  __shared__ __align__(16) bf16_t sWT[NPAD * KPAD];                 // 8 KB
  __shared__ __align__(16) bf16_t sUT[NPAD * KPAD];                 // 8 KB
  __shared__ float  sBias[NPAD];                                    // 512 B
  __shared__ __align__(16) float sHT[WAVES][ROWS_PER_WAVE * KPAD];  // 2 KB/wave

  const int tid  = threadIdx.x;
  const int wave = tid >> 5;           // wave32
  const int lane = tid & 31;

  // ---- cooperative weight staging (tiny: 16.5 KB total) ----
  for (int idx = tid; idx < NPAD * KPAD; idx += blockDim.x) {
    const int n = idx >> 5;            // KPAD == 32
    const int k = idx & 31;
    const int g = n >> 5;
    const int h = n & 31;
    const float* wg = (g == 0) ? wf : (g == 1) ? wi : (g == 2) ? wo : wc;
    const float* ug = (g == 0) ? uf : (g == 1) ? ui : (g == 2) ? uo : uc;
    const float wv = (h < H_HID && k < F_IN)  ? wg[k * H_HID + h] : 0.0f;
    const float uv = (h < H_HID && k < H_HID) ? ug[k * H_HID + h] : 0.0f;
    sWT[idx] = (bf16_t)wv;
    sUT[idx] = (bf16_t)uv;
  }
  for (int n = tid; n < NPAD; n += blockDim.x) {
    const int g = n >> 5, h = n & 31;
    const float* bg = (g == 0) ? bfp : (g == 1) ? bip : (g == 2) ? bop : bcp;
    sBias[n] = (h < H_HID) ? bg[h] : 0.0f;
  }
  __syncthreads();

  const int b0 = (blockIdx.x * WAVES + wave) * ROWS_PER_WAVE;
  if (b0 >= Btot) return;              // wave-uniform exit

  const int  nlo    = lane & 15;
  const bool hiHalf = (lane & 16) != 0;
  const int  mbase  = hiHalf ? 8 : 0;
  const int  kbA    = hiHalf ? 8 : 0;  // A-tile K base for this lane group

  // ---- resident U B-tiles only (reused 5x); W tiles re-read from LDS ----
  v16bf Ut[8];
#pragma unroll
  for (int nt = 0; nt < 8; ++nt)
    Ut[nt] = load_btile(sUT, nt * 16 + nlo, hiHalf ? 8 : 0);

  float biasv[8];
#pragma unroll
  for (int nt = 0; nt < 8; ++nt) biasv[nt] = sBias[nt * 16 + nlo];

  // ---- branchless x fetch: 8 x float2 per lane per timestep ----
  const float* xbase = x + (size_t)(b0 + nlo) * T_LEN * F_IN;
  float2 xv[8], xn[8];
  {
#pragma unroll
    for (int j = 0; j < 8; ++j) {
      const int k  = (j < 4) ? (kbA + 2 * j) : (kbA + 16 + 2 * (j - 4));
      const bool ok = (k < F_IN);                 // only k==30 (hi half) fails
      const float2 v = *(const float2*)(xbase + (ok ? k : 0));
      xv[j].x = ok ? v.x : 0.0f;
      xv[j].y = ok ? v.y : 0.0f;
    }
  }

  v8f   cst[2] = { {}, {} };           // cell state, D-layout (2 N-tiles)
  v16bf hA     = {};                   // h_{t-1} in A-layout (zero at t=0)
  float* hbuf  = sHT[wave];            // wave-private transpose buffer

#pragma unroll 1
  for (int t = 0; t < T_LEN; ++t) {
    // ---- pack current x into A-tile (bf16) ----
    v16bf aX;
#pragma unroll
    for (int j = 0; j < 4; ++j) {
      aX[2 * j]     = (bf16_t)xv[j].x;
      aX[2 * j + 1] = (bf16_t)xv[j].y;
      aX[8 + 2 * j]     = (bf16_t)xv[4 + j].x;
      aX[8 + 2 * j + 1] = (bf16_t)xv[4 + j].y;
    }

    // ---- prefetch next timestep's x (clamped t => always-valid loads) ----
    {
      const int tn = (t + 1 < T_LEN) ? (t + 1) : t;
      const float* xrow = xbase + tn * F_IN;
#pragma unroll
      for (int j = 0; j < 8; ++j) {
        const int k  = (j < 4) ? (kbA + 2 * j) : (kbA + 16 + 2 * (j - 4));
        const bool ok = (k < F_IN);
        const float2 v = *(const float2*)(xrow + (ok ? k : 0));
        xn[j].x = ok ? v.x : 0.0f;
        xn[j].y = ok ? v.y : 0.0f;
      }
    }

    // ---- two column-halves; gate g lives in tile 2g+half ----
    v8f hsto[2];
#pragma unroll
    for (int half = 0; half < 2; ++half) {
      v8f acc[4];
#pragma unroll
      for (int g = 0; g < 4; ++g) {
        const int nt = 2 * g + half;
        const v16bf Wt = load_btile(sWT, nt * 16 + nlo, hiHalf ? 8 : 0);
        v8f a;
#pragma unroll
        for (int v = 0; v < 8; ++v) a[v] = biasv[nt];
        a = __builtin_amdgcn_wmma_f32_16x16x32_bf16(false, aX, false, Wt,
                                                    (short)0, a, false, false);
        a = __builtin_amdgcn_wmma_f32_16x16x32_bf16(false, hA, false, Ut[nt],
                                                    (short)0, a, false, false);
        acc[g] = a;
      }
      const v8f ft = sig8(acc[0]);
      const v8f it = sig8(acc[1]);
      const v8f ot = sig8(acc[2]);
      const v8f ch = tanh8(acc[3]);
      v8f cc = cst[half];
      cc = ft * cc + it * ch;
      cst[half] = cc;
      hsto[half] = ot * tanh8(cc);
    }

    // ---- write h_t (D-layout -> [B,T,H]); half-wave rows are 64B contiguous ----
#pragma unroll
    for (int j = 0; j < 2; ++j) {
      const int col = j * 16 + nlo;
      if (col < H_HID) {
#pragma unroll
        for (int v = 0; v < 8; ++v) {
          out[((size_t)(b0 + mbase + v) * T_LEN + t) * H_HID + col] = hsto[j][v];
        }
      }
    }

    // ---- D->A transpose of h through wave-private LDS (no block barrier) ----
    if (t + 1 < T_LEN) {
#pragma unroll
      for (int j = 0; j < 2; ++j) {
        const int col = j * 16 + nlo;
#pragma unroll
        for (int v = 0; v < 8; ++v)
          hbuf[(mbase + v) * KPAD + col] = (col < H_HID) ? hsto[j][v] : 0.0f;
      }
      __builtin_amdgcn_wave_barrier();   // keep DS store->load order in schedule
      const float* hr = hbuf + nlo * KPAD + kbA;
      const float4 a0 = *(const float4*)(hr);
      const float4 a1 = *(const float4*)(hr + 4);
      const float4 a2 = *(const float4*)(hr + 16);
      const float4 a3 = *(const float4*)(hr + 20);
      hA[0]  = (bf16_t)a0.x; hA[1]  = (bf16_t)a0.y;
      hA[2]  = (bf16_t)a0.z; hA[3]  = (bf16_t)a0.w;
      hA[4]  = (bf16_t)a1.x; hA[5]  = (bf16_t)a1.y;
      hA[6]  = (bf16_t)a1.z; hA[7]  = (bf16_t)a1.w;
      hA[8]  = (bf16_t)a2.x; hA[9]  = (bf16_t)a2.y;
      hA[10] = (bf16_t)a2.z; hA[11] = (bf16_t)a2.w;
      hA[12] = (bf16_t)a3.x; hA[13] = (bf16_t)a3.y;
      hA[14] = (bf16_t)a3.z; hA[15] = (bf16_t)a3.w;
      __builtin_amdgcn_wave_barrier();   // reads done before next iter's stores
    }

    // rotate x prefetch
#pragma unroll
    for (int j = 0; j < 8; ++j) xv[j] = xn[j];
  }
}

extern "C" void kernel_launch(void* const* d_in, const int* in_sizes, int n_in,
                              void* d_out, int out_size, void* d_ws, size_t ws_size,
                              hipStream_t stream) {
  (void)n_in; (void)out_size; (void)d_ws; (void)ws_size;
  const float* x  = (const float*)d_in[0];
  const float* wf = (const float*)d_in[1];
  const float* wi = (const float*)d_in[2];
  const float* wo = (const float*)d_in[3];
  const float* wc = (const float*)d_in[4];
  const float* uf = (const float*)d_in[5];
  const float* ui = (const float*)d_in[6];
  const float* uo = (const float*)d_in[7];
  const float* uc = (const float*)d_in[8];
  const float* bf = (const float*)d_in[9];
  const float* bi = (const float*)d_in[10];
  const float* bo = (const float*)d_in[11];
  const float* bc = (const float*)d_in[12];
  float* out = (float*)d_out;

  const int Btot   = in_sizes[0] / (T_LEN * F_IN);
  const int blocks = (Btot + ROWS_PER_BLOCK - 1) / ROWS_PER_BLOCK;

  lstm_fused_wmma<<<blocks, WAVES * 32, 0, stream>>>(
      x, wf, wi, wo, wc, uf, ui, uo, uc, bf, bi, bo, bc, out, Btot);
}
